// BitLinearInf_35622458753042
// MI455X (gfx1250) — compile-verified, matrix operand order checked by reading, not run on hardware
//
#include <hip/hip_runtime.h>
#include <stdint.h>

typedef __attribute__((ext_vector_type(16))) _Float16 v16h;
typedef __attribute__((ext_vector_type(8)))  float    v8f;

#define N_ROWS  8192
#define IN_DIM  4096
#define OUT_DIM 4096
#define BM 128
#define BN 128
#define BK 64
#define LDSS 72   // padded LDS row stride in halves (144B -> conflict-free b128 reads)

// ---------------------------------------------------------------------------
// Kernel 0: unpack 1-bit weights into f16 {+1,-1}.  bit=0 -> +1 (0x3C00),
// bit=1 -> -1 (0xBC00).  One packed byte -> one uint4 (8 halves).
// ---------------------------------------------------------------------------
__global__ __launch_bounds__(256) void unpack_weights(
    const int* __restrict__ wpk,      // [OUT, IN/8] low byte used
    uint32_t* __restrict__ wf)        // [OUT, IN] halves as uint32 pairs
{
    const int idx = blockIdx.x * 256 + threadIdx.x;     // 0 .. OUT*IN/8-1
    const uint32_t b = (uint32_t)wpk[idx];
    const uint32_t one2 = 0x3C003C00u;
    uint4 o;
    o.x = one2 | ((b        & 1u) << 15) | ((b        & 2u) << 30);
    o.y = one2 | (((b >> 2) & 1u) << 15) | (((b >> 2) & 2u) << 30);
    o.z = one2 | (((b >> 4) & 1u) << 15) | (((b >> 4) & 2u) << 30);
    o.w = one2 | (((b >> 6) & 1u) << 15) | (((b >> 6) & 2u) << 30);
    ((uint4*)wf)[idx] = o;
}

// ---------------------------------------------------------------------------
// Kernel 1: f16 WMMA GEMM with fused input_factor and weight_scale.
// Y[N,OUT] = (X * factor) . Wf^T * wscale
// ---------------------------------------------------------------------------
__global__ __launch_bounds__(256) void bitlinear_gemm(
    const float*    __restrict__ X,       // [N, IN] fp32
    const uint32_t* __restrict__ Wf,      // [OUT, IN] f16 (as uint32 pairs)
    const float*    __restrict__ factor,  // [IN]
    const float*    __restrict__ wscale,  // [OUT]
    float*          __restrict__ Y)       // [N, OUT]
{
    __shared__ __align__(16) _Float16 sA[2][BM * LDSS];
    __shared__ __align__(16) _Float16 sB[2][BN * LDSS];

    const int tid  = threadIdx.x;
    const int lane = tid & 31;
    const int wave = tid >> 5;            // 0..7
    const int wm   = wave & 1;            // 2 wave-rows  (64 rows each)
    const int wn   = wave >> 1;           // 4 wave-cols  (32 cols each)

    const int blockM = blockIdx.y * BM;
    const int blockN = blockIdx.x * BN;

    // Global->LDS staging maps (fully coalesced)
    const int a_c4 = tid & 15;            // float4 column  (col = a_c4*4)
    const int a_r0 = tid >> 4;            // row base, +16 per pass (8 passes)
    const int b_c8 = tid & 7;             // 16B chunk      (halfcol = b_c8*8)
    const int b_r0 = tid >> 3;            // row base, +32 per pass (4 passes)

    float4 aReg[8];
    uint4  bReg[4];
    float4 fReg;

    v8f acc[4][2];
    #pragma unroll
    for (int i = 0; i < 4; ++i)
        #pragma unroll
        for (int j = 0; j < 2; ++j)
            #pragma unroll
            for (int e = 0; e < 8; ++e) acc[i][j][e] = 0.0f;

    auto loadGlobal = [&](int k0) {
        fReg = *(const float4*)(factor + k0 + a_c4 * 4);
        #pragma unroll
        for (int p = 0; p < 8; ++p) {
            const int r = a_r0 + p * 16;
            aReg[p] = *(const float4*)(X + (size_t)(blockM + r) * IN_DIM + k0 + a_c4 * 4);
        }
        #pragma unroll
        for (int p = 0; p < 4; ++p) {
            const int r = b_r0 + p * 32;
            bReg[p] = *(const uint4*)(Wf + (size_t)(blockN + r) * (IN_DIM / 2)
                                         + (k0 >> 1) + b_c8 * 4);
        }
    };

    auto storeLDS = [&](int buf) {
        union { _Float16 h[4]; uint2 u; } pk;
        #pragma unroll
        for (int p = 0; p < 8; ++p) {
            const int r = a_r0 + p * 16;
            pk.h[0] = (_Float16)(aReg[p].x * fReg.x);
            pk.h[1] = (_Float16)(aReg[p].y * fReg.y);
            pk.h[2] = (_Float16)(aReg[p].z * fReg.z);
            pk.h[3] = (_Float16)(aReg[p].w * fReg.w);
            *(uint2*)(&sA[buf][r * LDSS + a_c4 * 4]) = pk.u;
        }
        #pragma unroll
        for (int p = 0; p < 4; ++p) {
            const int r = b_r0 + p * 32;
            *(uint4*)(&sB[buf][r * LDSS + b_c8 * 8]) = bReg[p];
        }
    };

    union Frag { v16h v; uint4 q[2]; };
    // CDNA5 16-bit operand layout: lane L holds row (L&15), K-chunks
    // [kc, kc+8) and [kc+16, kc+24) with kc = (L>>4)*8  -> two b128 reads.
    auto loadFrag = [&](const _Float16* base, int row, int kk) -> v16h {
        Frag f;
        const int m  = lane & 15;
        const int kc = (lane >> 4) * 8;
        const _Float16* p = base + (row + m) * LDSS + kk + kc;
        f.q[0] = *(const uint4*)(p);
        f.q[1] = *(const uint4*)(p + 16);
        return f.v;
    };

    auto compute = [&](int buf) {
        const _Float16* Ab = sA[buf];
        const _Float16* Bb = sB[buf];
        #pragma unroll
        for (int kk = 0; kk < BK; kk += 32) {
            v16h bfr[2];
            #pragma unroll
            for (int j = 0; j < 2; ++j)
                bfr[j] = loadFrag(Bb, wn * 32 + j * 16, kk);
            #pragma unroll
            for (int i = 0; i < 4; ++i) {
                const v16h afr = loadFrag(Ab, wm * 64 + i * 16, kk);
                #pragma unroll
                for (int j = 0; j < 2; ++j)
                    acc[i][j] = __builtin_amdgcn_wmma_f32_16x16x32_f16(
                        false, afr, false, bfr[j], (short)0, acc[i][j], false, false);
            }
        }
    };

    // Pipelined main loop: prefetch next tile to registers while computing.
    loadGlobal(0);
    storeLDS(0);
    __syncthreads();
    const int nsteps = IN_DIM / BK;   // 64
    for (int s = 0; s < nsteps; ++s) {
        const int buf = s & 1;
        if (s + 1 < nsteps) loadGlobal((s + 1) * BK);
        compute(buf);
        if (s + 1 < nsteps) storeLDS(buf ^ 1);
        __syncthreads();
    }

    // Epilogue: apply weight_scale, write Y.  C layout: col = lane&15,
    // row = vgpr + 8*(lane>>4).
    #pragma unroll
    for (int j = 0; j < 2; ++j) {
        const int col = blockN + wn * 32 + j * 16 + (lane & 15);
        const float sc = wscale[col];
        #pragma unroll
        for (int i = 0; i < 4; ++i) {
            const int rbase = blockM + wm * 64 + i * 16 + (lane >> 4) * 8;
            #pragma unroll
            for (int r = 0; r < 8; ++r)
                Y[(size_t)(rbase + r) * OUT_DIM + col] = acc[i][j][r] * sc;
        }
    }
}

// ---------------------------------------------------------------------------
// Kernel 2: per-row LayerNorm (no affine) + bias.  One block per row,
// deterministic shared-memory tree reduction.
// ---------------------------------------------------------------------------
__global__ __launch_bounds__(256) void layernorm_bias(
    float* __restrict__ Y, const float* __restrict__ bias)
{
    __shared__ float ssum[256];
    __shared__ float ssq[256];
    const int t = threadIdx.x;
    float* yr = Y + (size_t)blockIdx.x * OUT_DIM;

    float4 v[4];
    float sum = 0.0f, sq = 0.0f;
    #pragma unroll
    for (int i = 0; i < 4; ++i) {
        v[i] = *(const float4*)(yr + t * 4 + i * 1024);
        sum += v[i].x + v[i].y + v[i].z + v[i].w;
        sq  += v[i].x * v[i].x + v[i].y * v[i].y + v[i].z * v[i].z + v[i].w * v[i].w;
    }
    ssum[t] = sum; ssq[t] = sq;
    __syncthreads();
    #pragma unroll
    for (int off = 128; off > 0; off >>= 1) {
        if (t < off) { ssum[t] += ssum[t + off]; ssq[t] += ssq[t + off]; }
        __syncthreads();
    }
    const float mean = ssum[0] * (1.0f / OUT_DIM);
    const float var  = ssq[0] * (1.0f / OUT_DIM) - mean * mean;
    const float rstd = rsqrtf(var + 1e-5f);

    #pragma unroll
    for (int i = 0; i < 4; ++i) {
        const float4 b4 = *(const float4*)(bias + t * 4 + i * 1024);
        float4 o;
        o.x = (v[i].x - mean) * rstd + b4.x;
        o.y = (v[i].y - mean) * rstd + b4.y;
        o.z = (v[i].z - mean) * rstd + b4.z;
        o.w = (v[i].w - mean) * rstd + b4.w;
        *(float4*)(yr + t * 4 + i * 1024) = o;
    }
}

// ---------------------------------------------------------------------------
extern "C" void kernel_launch(void* const* d_in, const int* in_sizes, int n_in,
                              void* d_out, int out_size, void* d_ws, size_t ws_size,
                              hipStream_t stream) {
    const float* X      = (const float*)d_in[0];   // [N, IN]
    const int*   W      = (const int*)  d_in[1];   // [OUT, IN/8]
    const float* wscale = (const float*)d_in[2];   // [OUT]
    const float* factor = (const float*)d_in[3];   // [IN]
    const float* bias   = (const float*)d_in[4];   // [OUT]
    float*       Y      = (float*)d_out;           // [N, OUT]
    uint32_t*    Wf     = (uint32_t*)d_ws;         // 32 MB: OUT*IN f16

    // 1) unpack packed bits -> f16 +-1 matrix in workspace (L2-resident)
    unpack_weights<<<(OUT_DIM * IN_DIM / 8) / 256, 256, 0, stream>>>(W, Wf);

    // 2) WMMA GEMM; OUT-blocks fastest so a row band + full W share L2
    dim3 grid(OUT_DIM / BN, N_ROWS / BM);          // (32, 64)
    bitlinear_gemm<<<grid, 256, 0, stream>>>(X, Wf, factor, wscale, Y);

    // 3) per-row LayerNorm + bias (deterministic)
    layernorm_bias<<<N_ROWS, 256, 0, stream>>>(Y, bias);
}
